// SS3D_89326729822214
// MI455X (gfx1250) — compile-verified, matrix operand order checked by reading, not run on hardware
//
#include <hip/hip_runtime.h>

// ---------------- problem constants ----------------
constexpr int DMODEL = 128;
constexpr int DINNER = 256;
constexpr int DSTATE = 16;
constexpr int DTRANK = 8;
constexpr int KD     = 8;                 // scan directions
constexpr int TT = 8, HH = 24, WW = 24;
constexpr int L   = TT * HH * WW;         // 4608
constexpr int HWp = HH * WW;              // 576
constexpr int CPROJ = DTRANK + 2 * DSTATE; // 40 (dt | B | C)
constexpr int CL = 64;                    // scan chunk length
constexpr int NC = L / CL;                // 72 chunks

typedef __attribute__((ext_vector_type(16))) _Float16 v16h;
typedef __attribute__((ext_vector_type(8)))  float    v8f;

// 16-bit A/B WMMA lane element -> K offset within a 32-wide K chunk.
// half = lane>>4. e<8 -> half*8+e ; e>=8 -> half*8+16+(e-8)   (ISA 7.12.2)
__device__ __forceinline__ int kmap16(int e, int half) {
  return half * 8 + (e & 7) + ((e >> 3) << 4);
}

// -------- perm table: scan index i of direction k -> THW flat index --------
__global__ void k_perm(int* __restrict__ perm) {
  int idx = blockIdx.x * blockDim.x + threadIdx.x;
  if (idx >= KD * L) return;
  int k = idx / L, i = idx % L;
  int i0 = (k < 4) ? i : (L - 1 - i);     // flipped directions
  int kk = k & 3;
  int t, h, w;
  if (kk == 0)      { t = i0 / HWp;        h = (i0 % HWp) / WW; w = i0 % WW; }       // (T,H,W)
  else if (kk == 1) { t = i0 / HWp;        w = (i0 % HWp) / HH; h = i0 % HH; }       // (T,W,H)
  else if (kk == 2) { h = i0 / (WW * TT);  w = (i0 / TT) % WW;  t = i0 % TT; }       // (H,W,T)
  else              { w = i0 / (HH * TT);  h = (i0 / TT) % HH;  t = i0 % TT; }       // (W,H,T)
  perm[idx] = t * HWp + h * WW + w;
}

// -------- in_proj: xz[L,512] = x[L,128] @ in_proj_w[512,128]^T  (WMMA f16) ----
__global__ void k_inproj(const float* __restrict__ x, const float* __restrict__ w,
                         float* __restrict__ xz) {
  int wid  = (blockIdx.x * blockDim.x + threadIdx.x) >> 5;
  int lane = threadIdx.x & 31;
  constexpr int NT = (2 * DINNER) / 16;   // 32 column tiles
  int mt = wid / NT, nt = wid % NT;
  int m0 = mt * 16, n0 = nt * 16;
  int half = lane >> 4, r = lane & 15;
  int arow = m0 + r, bcol = n0 + r;
  v8f acc = {};
  for (int kc = 0; kc < DMODEL; kc += 32) {
    v16h a, b;
#pragma unroll
    for (int e = 0; e < 16; ++e) {
      int kk = kc + kmap16(e, half);
      a[e] = (_Float16)x[arow * DMODEL + kk];
      b[e] = (_Float16)w[bcol * DMODEL + kk];
    }
    acc = __builtin_amdgcn_wmma_f32_16x16x32_f16(false, a, false, b, (short)0, acc, false, false);
  }
#pragma unroll
  for (int v = 0; v < 8; ++v)
    xz[(m0 + v + half * 8) * (2 * DINNER) + n0 + r] = acc[v];
}

// -------- depthwise 3x3x3 conv + bias + SiLU; output channel-major xc[d,L] ----
// threads indexed j*256+d so the 27-tap xz reads are coalesced across d.
__global__ void k_conv(const float* __restrict__ xz, const float* __restrict__ cw,
                       const float* __restrict__ cb, float* __restrict__ xc) {
  int idx = blockIdx.x * blockDim.x + threadIdx.x;
  if (idx >= DINNER * L) return;
  int d = idx % DINNER, j = idx / DINNER;
  int t = j / HWp, h = (j % HWp) / WW, w = j % WW;
  float acc = cb[d];
  for (int dt = -1; dt <= 1; ++dt) {
    int tt = t + dt; if ((unsigned)tt >= (unsigned)TT) continue;
    for (int dh = -1; dh <= 1; ++dh) {
      int hh = h + dh; if ((unsigned)hh >= (unsigned)HH) continue;
      for (int dw = -1; dw <= 1; ++dw) {
        int w2 = w + dw; if ((unsigned)w2 >= (unsigned)WW) continue;
        acc += xz[(tt * HWp + hh * WW + w2) * (2 * DINNER) + d]
             * cw[d * 27 + (dt + 1) * 9 + (dh + 1) * 3 + (dw + 1)];
      }
    }
  }
  xc[d * L + j] = acc / (1.f + __expf(-acc));   // SiLU
}

// -------- x_dbl: dbc[k,L,40] = xs[k](L,256) @ x_proj_w[k](40,256)^T (WMMA) ----
__global__ void k_xdbl(const float* __restrict__ xc, const float* __restrict__ xpw,
                       const int* __restrict__ perm, float* __restrict__ dbc) {
  int wid  = (blockIdx.x * blockDim.x + threadIdx.x) >> 5;
  int lane = threadIdx.x & 31;
  constexpr int NT = 3, MT = L / 16;      // 40 cols padded to 48
  int nt = wid % NT;
  int mt = (wid / NT) % MT;
  int k  = wid / (NT * MT);
  int m0 = mt * 16, n0 = nt * 16;
  int half = lane >> 4, r = lane & 15;
  int pj   = perm[k * L + m0 + r];        // gather: xs[k,d,i] = xc[d, perm(i)]
  int bcol = n0 + r;
  v8f acc = {};
  for (int kc = 0; kc < DINNER; kc += 32) {
    v16h a, b;
#pragma unroll
    for (int e = 0; e < 16; ++e) {
      int kk = kc + kmap16(e, half);
      a[e] = (_Float16)xc[kk * L + pj];
      b[e] = (bcol < CPROJ) ? (_Float16)xpw[(k * CPROJ + bcol) * DINNER + kk]
                            : (_Float16)0.f;
    }
    acc = __builtin_amdgcn_wmma_f32_16x16x32_f16(false, a, false, b, (short)0, acc, false, false);
  }
#pragma unroll
  for (int v = 0; v < 8; ++v) {
    int col = n0 + r;
    if (col < CPROJ)
      dbc[(k * L + m0 + v + half * 8) * CPROJ + col] = acc[v];
  }
}

// -------- delta[k,d,L] = softplus(dts @ dt_w^T + dt_b) -----------------------
__global__ void k_delta(const float* __restrict__ dbc, const float* __restrict__ dtw,
                        const float* __restrict__ dtb, float* __restrict__ delta) {
  int idx = blockIdx.x * blockDim.x + threadIdx.x;
  if (idx >= KD * DINNER * L) return;
  int l = idx % L; int kd = idx / L;
  const float* dts = &dbc[(kd / DINNER * L + l) * CPROJ];
  const float* wv  = &dtw[(size_t)kd * DTRANK];
  float s = dtb[kd];
#pragma unroll
  for (int rr = 0; rr < DTRANK; ++rr) s += dts[rr] * wv[rr];
  delta[idx] = (s > 20.f) ? s : log1pf(__expf(s));
}

// ======== chunked associative scan: h <- dA*h + dB over L, split into NC=72
// chunks of CL=64.  (a1,b1)∘(a2,b2) = (a1*a2, a2*b1 + b2). =================

// Pass 1: per (chain kd, state n, chunk c): local (aP = prod dA, bP = h|h0=0)
__global__ void k_scan1(const float* __restrict__ xc, const float* __restrict__ dbc,
                        const float* __restrict__ delta, const float* __restrict__ Alog,
                        const int* __restrict__ perm,
                        float* __restrict__ aP, float* __restrict__ bP) {
  int tid = blockIdx.x * blockDim.x + threadIdx.x;
  int grp = tid >> 4;                     // (kd, c), 0 .. 2048*NC-1
  int n   = tid & 15;
  if (grp >= KD * DINNER * NC) return;
  int c  = grp % NC;
  int kd = grp / NC;
  int k = kd / DINNER, d = kd % DINNER;
  float An = -__expf(Alog[kd * DSTATE + n]);
  const float* dptr = delta + (size_t)kd * L;
  const int*   pptr = perm + k * L;
  const float* xrow = xc + (size_t)d * L;
  float ap = 1.f, bp = 0.f;
  int i0 = c * CL;
  for (int s = 0; s < CL; ++s) {
    int i = i0 + s;
    __builtin_prefetch(dptr + i + 16, 0, 0);
    int j     = pptr[i];
    float dlt = dptr[i];
    float u   = xrow[j];
    float Bn  = dbc[(size_t)(k * L + i) * CPROJ + DTRANK + n];
    float dA  = __expf(dlt * An);
    bp = dA * bp + (dlt * u) * Bn;
    ap *= dA;
  }
  size_t idx = ((size_t)kd * NC + c) * DSTATE + n;
  aP[idx] = ap;
  bP[idx] = bp;
}

// Pass 2: per (chain kd, state n): sequential carry across NC chunks;
// overwrite bP[c] with the EXCLUSIVE carry (h at chunk entry).
__global__ void k_scan2(const float* __restrict__ aP, float* __restrict__ bP) {
  int tid = blockIdx.x * blockDim.x + threadIdx.x;
  if (tid >= KD * DINNER * DSTATE) return;
  int n  = tid & 15;
  int kd = tid >> 4;
  float h = 0.f;
  for (int c = 0; c < NC; ++c) {
    size_t idx = ((size_t)kd * NC + c) * DSTATE + n;
    float a = aP[idx], b = bP[idx];
    bP[idx] = h;                 // exclusive carry-in for chunk c
    h = a * h + b;
  }
}

// Pass 3: per (chain kd, chunk c): replay chunk from carry-in, emit outputs.
__global__ void k_scan3(const float* __restrict__ xc, const float* __restrict__ dbc,
                        const float* __restrict__ delta, const float* __restrict__ Alog,
                        const float* __restrict__ Dsv, const int* __restrict__ perm,
                        const float* __restrict__ bP, float* __restrict__ ysum) {
  int tid = blockIdx.x * blockDim.x + threadIdx.x;
  int grp = tid >> 4;
  int n   = tid & 15;
  if (grp >= KD * DINNER * NC) return;
  int c  = grp % NC;
  int kd = grp / NC;
  int k = kd / DINNER, d = kd % DINNER;
  float An = -__expf(Alog[kd * DSTATE + n]);
  float Dc = Dsv[kd];
  const float* dptr = delta + (size_t)kd * L;
  const int*   pptr = perm + k * L;
  const float* xrow = xc + (size_t)d * L;
  float h = bP[((size_t)kd * NC + c) * DSTATE + n];   // carry-in
  int i0 = c * CL;
  for (int s = 0; s < CL; ++s) {
    int i = i0 + s;
    __builtin_prefetch(dptr + i + 16, 0, 0);
    int j     = pptr[i];
    float dlt = dptr[i];
    float u   = xrow[j];
    const float* bc = dbc + (size_t)(k * L + i) * CPROJ;
    float Bn = bc[DTRANK + n];
    float Cn = bc[DTRANK + DSTATE + n];
    h = __expf(dlt * An) * h + (dlt * u) * Bn;
    float y = h * Cn;                       // reduce over 16 states
    y += __shfl_xor(y, 1, 16);
    y += __shfl_xor(y, 2, 16);
    y += __shfl_xor(y, 4, 16);
    y += __shfl_xor(y, 8, 16);
    if (n == 0) atomicAdd(&ysum[j * DINNER + d], y + Dc * u);  // merge 8 dirs
  }
}

// -------- LayerNorm over 256 + SiLU(z) gate ----------------------------------
__global__ void k_lngate(const float* __restrict__ ysum, const float* __restrict__ xz,
                         const float* __restrict__ lng, const float* __restrict__ lnb,
                         float* __restrict__ yg) {
  int j = blockIdx.x, d = threadIdx.x;
  __shared__ float s1[DINNER], s2[DINNER];
  float y = ysum[j * DINNER + d];
  s1[d] = y; s2[d] = y * y;
  __syncthreads();
  for (int off = DINNER / 2; off > 0; off >>= 1) {
    if (d < off) { s1[d] += s1[d + off]; s2[d] += s2[d + off]; }
    __syncthreads();
  }
  float mu  = s1[0] * (1.f / DINNER);
  float var = s2[0] * (1.f / DINNER) - mu * mu;
  float yn  = (y - mu) * rsqrtf(var + 1e-5f) * lng[d] + lnb[d];
  float z   = xz[j * (2 * DINNER) + DINNER + d];
  yg[j * DINNER + d] = yn * (z / (1.f + __expf(-z)));
}

// -------- out_proj: out[L,128] = yg[L,256] @ out_proj_w[128,256]^T (WMMA) ----
__global__ void k_outproj(const float* __restrict__ yg, const float* __restrict__ opw,
                          float* __restrict__ out) {
  int wid  = (blockIdx.x * blockDim.x + threadIdx.x) >> 5;
  int lane = threadIdx.x & 31;
  constexpr int NT = DMODEL / 16;         // 8
  int mt = wid / NT, nt = wid % NT;
  int m0 = mt * 16, n0 = nt * 16;
  int half = lane >> 4, r = lane & 15;
  v8f acc = {};
  for (int kc = 0; kc < DINNER; kc += 32) {
    v16h a, b;
#pragma unroll
    for (int e = 0; e < 16; ++e) {
      int kk = kc + kmap16(e, half);
      a[e] = (_Float16)yg[(m0 + r) * DINNER + kk];
      b[e] = (_Float16)opw[(n0 + r) * DINNER + kk];
    }
    acc = __builtin_amdgcn_wmma_f32_16x16x32_f16(false, a, false, b, (short)0, acc, false, false);
  }
#pragma unroll
  for (int v = 0; v < 8; ++v)
    out[(m0 + v + half * 8) * DMODEL + n0 + r] = acc[v];
}

extern "C" void kernel_launch(void* const* d_in, const int* in_sizes, int n_in,
                              void* d_out, int out_size, void* d_ws, size_t ws_size,
                              hipStream_t stream) {
  (void)in_sizes; (void)n_in; (void)out_size; (void)ws_size;
  const float* x    = (const float*)d_in[0];
  const float* ipw  = (const float*)d_in[1];
  const float* cw   = (const float*)d_in[2];
  const float* cb   = (const float*)d_in[3];
  const float* xpw  = (const float*)d_in[4];
  const float* dtw  = (const float*)d_in[5];
  const float* dtb  = (const float*)d_in[6];
  const float* Alog = (const float*)d_in[7];
  const float* Dsv  = (const float*)d_in[8];
  const float* lng  = (const float*)d_in[9];
  const float* lnb  = (const float*)d_in[10];
  const float* opw  = (const float*)d_in[11];
  float* out = (float*)d_out;

  char* p = (char*)d_ws;
  auto take = [&](size_t bytes) { char* r = p; p += (bytes + 255) & ~(size_t)255; return r; };
  float* xz    = (float*)take((size_t)L * 2 * DINNER * sizeof(float));        // 9.4 MB
  float* xc    = (float*)take((size_t)DINNER * L * sizeof(float));            // 4.7 MB
  float* dbc   = (float*)take((size_t)KD * L * CPROJ * sizeof(float));        // 5.9 MB
  float* delta = (float*)take((size_t)KD * DINNER * L * sizeof(float));       // 37.7 MB
  int*   perm  = (int*)  take((size_t)KD * L * sizeof(int));                  // 0.15 MB
  float* ysum  = (float*)take((size_t)L * DINNER * sizeof(float));            // 4.7 MB
  float* yg    = (float*)take((size_t)L * DINNER * sizeof(float));            // 4.7 MB
  float* aP    = (float*)take((size_t)KD * DINNER * NC * DSTATE * sizeof(float)); // 9.4 MB
  float* bP    = (float*)take((size_t)KD * DINNER * NC * DSTATE * sizeof(float)); // 9.4 MB

  hipMemsetAsync(ysum, 0, (size_t)L * DINNER * sizeof(float), stream);

  k_perm   <<<(KD * L + 255) / 256, 256, 0, stream>>>(perm);
  k_inproj <<<(L / 16) * ((2 * DINNER) / 16) * 32 / 256, 256, 0, stream>>>(x, ipw, xz);
  k_conv   <<<(DINNER * L + 255) / 256, 256, 0, stream>>>(xz, cw, cb, xc);
  k_xdbl   <<<KD * (L / 16) * 3 * 32 / 256, 256, 0, stream>>>(xc, xpw, perm, dbc);
  k_delta  <<<(KD * DINNER * L + 255) / 256, 256, 0, stream>>>(dbc, dtw, dtb, delta);
  k_scan1  <<<KD * DINNER * NC * 16 / 256, 256, 0, stream>>>(xc, dbc, delta, Alog, perm, aP, bP);
  k_scan2  <<<(KD * DINNER * DSTATE + 255) / 256, 256, 0, stream>>>(aP, bP);
  k_scan3  <<<KD * DINNER * NC * 16 / 256, 256, 0, stream>>>(xc, dbc, delta, Alog, Dsv, perm, bP, ysum);
  k_lngate <<<L, DINNER, 0, stream>>>(ysum, xz, lng, lnb, yg);
  k_outproj<<<(L / 16) * (DMODEL / 16) * 32 / 256, 256, 0, stream>>>(yg, opw, out);
}
